// TransformerMoE_62560493633923
// MI455X (gfx1250) — matrix operand order repaired and checked by reading, not compiled, not run on hardware
//
#include <hip/hip_runtime.h>
#include <math.h>

#define D_MODEL 1024
#define D_FF    4096
#define NE      8
#define NTOK    2048              // SEQ * BATCH

#define MTA     32                // GEMM1 tokens per block (2 x 16 tiles)
#define MTB     64                // GEMM2 tokens per block (4 x 16 tiles)
#define WAVES   8
#define N1F     64                // GEMM1 cols per wave (4 n-tiles)
#define F_CHUNK (N1F * WAVES)     // 512
#define N2      32                // GEMM2 cols per wave (2 n-tiles)
#define DBLK    (N2 * WAVES)      // 256 d-cols per block
#define MAX_SLOTS 4608            // sum ceil64(cnt_e) <= 2*T + 8*63 -> 4600

typedef __bf16 bf16_t;
typedef __attribute__((ext_vector_type(16))) __bf16 v16bf;
typedef __attribute__((ext_vector_type(8)))  __bf16 v8bf;
typedef __attribute__((ext_vector_type(4)))  __bf16 v4bf;
typedef __attribute__((ext_vector_type(8)))  float  v8f;
typedef __attribute__((ext_vector_type(4)))  float  v4f;

// ---------------------------------------------------------------------------
// x fp32 -> bf16 (A-fragments then load 16B bf16 chunks, no inner-loop cvt)
// ---------------------------------------------------------------------------
__global__ __launch_bounds__(256) void cvt_x(const float* __restrict__ x,
                                             bf16_t* __restrict__ xb) {
  const int i = blockIdx.x * blockDim.x + threadIdx.x;   // 4 elements each
  const v4f f = ((const v4f*)x)[i];
  v4bf o;
#pragma unroll
  for (int j = 0; j < 4; ++j) o[j] = (bf16_t)f[j];
  ((v4bf*)xb)[i] = o;
}

// ---------------------------------------------------------------------------
// Tiled transpose+convert: src[R][C] fp32 -> dst[C][R] bf16 (per-expert slab).
// Both global read and write are coalesced; LDS does the corner turn.
// ---------------------------------------------------------------------------
__global__ __launch_bounds__(256) void transpose_cvt(
    const float* __restrict__ src, bf16_t* __restrict__ dst, int R, int C) {
  __shared__ bf16_t tile[64][72];                 // [col][row], padded
  const size_t slab = (size_t)R * C * blockIdx.z;
  src += slab;  dst += slab;
  const int C0 = blockIdx.x * 64, R0 = blockIdx.y * 64;
  const int tr  = threadIdx.x >> 4;               // 0..15
  const int tc4 = (threadIdx.x & 15) * 4;         // 0..60
#pragma unroll
  for (int it = 0; it < 4; ++it) {
    const int r = it * 16 + tr;
    const v4f f = *(const v4f*)&src[(size_t)(R0 + r) * C + C0 + tc4];
#pragma unroll
    for (int j = 0; j < 4; ++j) tile[tc4 + j][r] = (bf16_t)f[j];
  }
  __syncthreads();
#pragma unroll
  for (int it = 0; it < 4; ++it) {
    const int c = it * 16 + tr;                   // output row = source col
    v4bf o;
#pragma unroll
    for (int j = 0; j < 4; ++j) o[j] = tile[c][tc4 + j];
    *(v4bf*)&dst[(size_t)(C0 + c) * R + R0 + tc4] = o;
  }
}

// ---------------------------------------------------------------------------
// Gating: one wave32 per token. Records per-expert lists (for GEMM gather)
// and per-token (expert, position, prob) pairs (for the final combine).
// ---------------------------------------------------------------------------
__global__ __launch_bounds__(256) void moe_gate(
    const float* __restrict__ x, const float* __restrict__ gw,
    const float* __restrict__ gb, int* __restrict__ counts,
    int* __restrict__ tokIdx, int* __restrict__ slotE,
    int* __restrict__ slotS, float* __restrict__ slotW) {
  const int lane = threadIdx.x & 31;
  const int t = blockIdx.x * (blockDim.x >> 5) + (threadIdx.x >> 5);
  if (t >= NTOK) return;

  const float* xr = x + (size_t)t * D_MODEL;
  float acc[NE];
#pragma unroll
  for (int e = 0; e < NE; ++e) acc[e] = 0.f;
  for (int d = lane; d < D_MODEL; d += 32) {
    const float xv = xr[d];
    const float* g = gw + (size_t)d * NE;
#pragma unroll
    for (int e = 0; e < NE; ++e) acc[e] += xv * g[e];
  }
#pragma unroll
  for (int e = 0; e < NE; ++e) {
#pragma unroll
    for (int off = 16; off > 0; off >>= 1)
      acc[e] += __shfl_xor(acc[e], off, 32);
  }
  if (lane == 0) {
    float sc[NE];
    int i0 = 0; float v0 = acc[0] + gb[0];
    sc[0] = v0;
#pragma unroll
    for (int e = 1; e < NE; ++e) {
      sc[e] = acc[e] + gb[e];
      if (sc[e] > v0) { v0 = sc[e]; i0 = e; }
    }
    int i1 = (i0 == 0) ? 1 : 0; float v1 = sc[i1];
#pragma unroll
    for (int e = 0; e < NE; ++e)
      if (e != i0 && sc[e] > v1) { v1 = sc[e]; i1 = e; }

    const float p0 = 1.f / (1.f + __expf(v1 - v0));
    const float p1 = 1.f - p0;
    const int s0 = atomicAdd(&counts[i0], 1);
    tokIdx[i0 * NTOK + s0] = t;
    const int s1 = atomicAdd(&counts[i1], 1);
    tokIdx[i1 * NTOK + s1] = t;
    slotE[2 * t] = i0;  slotS[2 * t] = s0;  slotW[2 * t] = p0;
    slotE[2 * t + 1] = i1;  slotS[2 * t + 1] = s1;  slotW[2 * t + 1] = p1;
  }
}

// Exclusive scan of per-expert counts, padded to 64-row tiles.
__global__ void scan_counts(const int* __restrict__ counts,
                            int* __restrict__ offsets) {
  if (threadIdx.x == 0 && blockIdx.x == 0) {
    int s = 0;
#pragma unroll
    for (int e = 0; e < NE; ++e) {
      offsets[e] = s;
      s += (counts[e] + 63) & ~63;
    }
  }
}

// ---------------------------------------------------------------------------
// GEMM1: h[slot, f] = relu(x[tok] @ W1[e] + b1[e]).  W1t is [e][F][D] bf16,
// so B-fragments are two contiguous b128 loads (K runs along the row).
// ---------------------------------------------------------------------------
__global__ __launch_bounds__(256) void moe_ffn1(
    const bf16_t* __restrict__ xb, const bf16_t* __restrict__ W1t,
    const float* __restrict__ b1, const int* __restrict__ counts,
    const int* __restrict__ tokIdx, const int* __restrict__ offsets,
    bf16_t* __restrict__ hbuf) {
  const int e = blockIdx.y;
  const int tileBase = blockIdx.x * MTA;
  const int cnt = counts[e];
  if (tileBase >= cnt) return;                    // uniform
  int tileCnt = cnt - tileBase; if (tileCnt > MTA) tileCnt = MTA;

  const int lane = threadIdx.x & 31;
  const int wave = threadIdx.x >> 5;
  const int mrow = lane & 15;
  const int hi16 = lane >> 4;
  const int b8   = hi16 * 8;
  const int b16  = hi16 * 16;

  const bf16_t* xrow[2];
#pragma unroll
  for (int m = 0; m < 2; ++m) {
    const int r = m * 16 + mrow;
    const int tk = (r < tileCnt) ? tokIdx[e * NTOK + tileBase + r] : 0;
    xrow[m] = xb + (size_t)tk * D_MODEL;
  }
  const bf16_t* W1te = W1t + (size_t)e * D_FF * D_MODEL;
  const int slotBase = offsets[e] + tileBase;

  for (int fc = 0; fc < D_FF; fc += F_CHUNK) {
    const bf16_t* brow[4];
#pragma unroll
    for (int n = 0; n < 4; ++n)
      brow[n] = W1te + (size_t)(fc + wave * N1F + n * 16 + mrow) * D_MODEL;

    v8f acc[2][4];
#pragma unroll
    for (int m = 0; m < 2; ++m)
#pragma unroll
      for (int n = 0; n < 4; ++n) acc[m][n] = (v8f){0,0,0,0,0,0,0,0};

#pragma unroll 2
    for (int k0 = 0; k0 < D_MODEL; k0 += 32) {
      v16bf a[2];
#pragma unroll
      for (int m = 0; m < 2; ++m) {
        const bf16_t* p = xrow[m] + k0 + b8;
        const v8bf lo = *(const v8bf*)(p);
        const v8bf hi = *(const v8bf*)(p + 16);
#pragma unroll
        for (int j = 0; j < 8; ++j) { a[m][j] = lo[j]; a[m][j + 8] = hi[j]; }
      }
#pragma unroll
      for (int n = 0; n < 4; ++n) {
        const bf16_t* p = brow[n] + k0 + b8;
        const v8bf blo = *(const v8bf*)(p);
        const v8bf bhi = *(const v8bf*)(p + 16);
        v16bf bfrag;
#pragma unroll
        for (int j = 0; j < 8; ++j) { bfrag[j] = blo[j]; bfrag[j + 8] = bhi[j]; }
#pragma unroll
        for (int m = 0; m < 2; ++m)
          acc[m][n] = __builtin_amdgcn_wmma_f32_16x16x32_bf16(
              false, a[m], false, bfrag, (short)0, acc[m][n], false, false);
      }
    }
    // bias + relu + bf16 store (lanes 0..15 write 16 consecutive halfs)
#pragma unroll
    for (int n = 0; n < 4; ++n) {
      const int fl = fc + wave * N1F + n * 16 + mrow;
      const float bias = b1[e * D_FF + fl];
#pragma unroll
      for (int m = 0; m < 2; ++m)
#pragma unroll
        for (int r = 0; r < 8; ++r) {
          const int tl = m * 16 + r + hi16 * 8;
          const float hv = acc[m][n][r] + bias;
          hbuf[(size_t)(slotBase + tl) * D_FF + fl] = (bf16_t)(hv > 0.f ? hv : 0.f);
        }
    }
  }
}

// ---------------------------------------------------------------------------
// GEMM2: ybuf[slot, d] = h[slot] @ W2[e].  W2t is [e][D][F] bf16; fragments
// on both sides are contiguous b128 loads.  Plain (non-atomic) stores.
// ---------------------------------------------------------------------------
__global__ __launch_bounds__(256) void moe_ffn2(
    const bf16_t* __restrict__ hbuf, const bf16_t* __restrict__ W2t,
    const int* __restrict__ counts, const int* __restrict__ offsets,
    float* __restrict__ ybuf) {
  const int e = blockIdx.z;
  const int tileBase = blockIdx.x * MTB;
  const int cnt = counts[e];
  if (tileBase >= cnt) return;                    // uniform

  const int lane = threadIdx.x & 31;
  const int wave = threadIdx.x >> 5;
  const int mrow = lane & 15;
  const int hi16 = lane >> 4;
  const int b8   = hi16 * 8;
  const int b16  = hi16 * 16;
  const int d0   = blockIdx.y * DBLK + wave * N2;

  const int slotBase = offsets[e] + tileBase;
  const bf16_t* hrow[4];
#pragma unroll
  for (int m = 0; m < 4; ++m)
    hrow[m] = hbuf + (size_t)(slotBase + m * 16 + mrow) * D_FF;
  const bf16_t* W2te = W2t + (size_t)e * D_FF * D_MODEL;
  const bf16_t* brow[2];
#pragma unroll
  for (int n = 0; n < 2; ++n)
    brow[n] = W2te + (size_t)(d0 + n * 16 + mrow) * D_FF;

  v8f acc[4][2];
#pragma unroll
  for (int m = 0; m < 4; ++m)
#pragma unroll
    for (int n = 0; n < 2; ++n) acc[m][n] = (v8f){0,0,0,0,0,0,0,0};

#pragma unroll 2
  for (int kf = 0; kf < D_FF; kf += 32) {
    v16bf a[4];
#pragma unroll
    for (int m = 0; m < 4; ++m) {
      const bf16_t* p = hrow[m] + kf + b8;
      const v8bf lo = *(const v8bf*)(p);
      const v8bf hi = *(const v8bf*)(p + 16);
#pragma unroll
      for (int j = 0; j < 8; ++j) { a[m][j] = lo[j]; a[m][j + 8] = hi[j]; }
    }
#pragma unroll
    for (int n = 0; n < 2; ++n) {
      const bf16_t* p = brow[n] + kf + b8;
      const v8bf blo = *(const v8bf*)(p);
      const v8bf bhi = *(const v8bf*)(p + 16);
      v16bf bfrag;
#pragma unroll
      for (int j = 0; j < 8; ++j) { bfrag[j] = blo[j]; bfrag[j + 8] = bhi[j]; }
#pragma unroll
      for (int m = 0; m < 4; ++m)
        acc[m][n] = __builtin_amdgcn_wmma_f32_16x16x32_bf16(
            false, a[m], false, bfrag, (short)0, acc[m][n], false, false);
    }
  }

  // unconditional stores; padded slots are never read by the combine pass
#pragma unroll
  for (int m = 0; m < 4; ++m)
#pragma unroll
    for (int n = 0; n < 2; ++n)
#pragma unroll
      for (int r = 0; r < 8; ++r) {
        const int tl = m * 16 + r + hi16 * 8;
        const int dcol = d0 + n * 16 + mrow;
        ybuf[(size_t)(slotBase + tl) * D_MODEL + dcol] = acc[m][n][r];
      }
}

// ---------------------------------------------------------------------------
// Combine: out[t] = sum_k w_k * (ybuf[slot_k] + b2[e_k]).  b128 in/out.
// ---------------------------------------------------------------------------
__global__ __launch_bounds__(256) void moe_combine(
    const float* __restrict__ ybuf, const float* __restrict__ b2,
    const int* __restrict__ slotE, const int* __restrict__ slotS,
    const float* __restrict__ slotW, const int* __restrict__ offsets,
    float* __restrict__ out) {
  const int idx = blockIdx.x * blockDim.x + threadIdx.x;  // 4 cols each
  const int t  = idx >> 8;                                // D_MODEL/4 = 256
  const int c4 = (idx & 255) * 4;
  const int e0 = slotE[2 * t],     e1 = slotE[2 * t + 1];
  const int s0 = offsets[e0] + slotS[2 * t];
  const int s1 = offsets[e1] + slotS[2 * t + 1];
  const float w0 = slotW[2 * t],   w1 = slotW[2 * t + 1];
  const v4f y0 = *(const v4f*)&ybuf[(size_t)s0 * D_MODEL + c4];
  const v4f y1 = *(const v4f*)&ybuf[(size_t)s1 * D_MODEL + c4];
  const v4f g0 = *(const v4f*)&b2[e0 * D_MODEL + c4];
  const v4f g1 = *(const v4f*)&b2[e1 * D_MODEL + c4];
  v4f o;
#pragma unroll
  for (int j = 0; j < 4; ++j) o[j] = w0 * (y0[j] + g0[j]) + w1 * (y1[j] + g1[j]);
  *(v4f*)&out[(size_t)t * D_MODEL + c4] = o;
}

// ---------------------------------------------------------------------------
extern "C" void kernel_launch(void* const* d_in, const int* in_sizes, int n_in,
                              void* d_out, int out_size, void* d_ws, size_t ws_size,
                              hipStream_t stream) {
  const float* x  = (const float*)d_in[0];
  const float* gw = (const float*)d_in[1];
  const float* gb = (const float*)d_in[2];
  const float* W1 = (const float*)d_in[3];
  const float* b1 = (const float*)d_in[4];
  const float* W2 = (const float*)d_in[5];
  const float* b2 = (const float*)d_in[6];
  float* out = (float*)d_out;

  // ws layout (bytes):
  char* w = (char*)d_ws;
  int*    counts  = (int*)(w);                          // 128
  int*    tokIdx  = (int*)(w + 128);                    // 64 KB
  int*    offsets = (int*)(w + 128 + 65536);            // 128
  int*    slotE   = (int*)(w + 65792);                  // 16 KB
  int*    slotS   = (int*)(w + 65792 + 16384);          // 16 KB
  float*  slotW   = (float*)(w + 65792 + 32768);        // 16 KB
  bf16_t* xb      = (bf16_t*)(w + 114944);              // 4 MB
  bf16_t* hbuf    = (bf16_t*)(w + 114944 + 4194304);    // 36 MB
  bf16_t* W1t     = (bf16_t*)(w + 114944 + 4194304 + (size_t)MAX_SLOTS * D_FF * 2);
  bf16_t* W2t     = W1t + (size_t)NE * D_FF * D_MODEL;  // 64 MB each
  float*  ybuf    = (float*)(W2t + (size_t)NE * D_FF * D_MODEL);  // 18 MB

  hipMemsetAsync(counts, 0, 128, stream);

  cvt_x<<<(NTOK * D_MODEL) / (4 * 256), 256, 0, stream>>>(x, xb);
  // W1: [E][D][F] -> W1t [E][F][D] ; W2: [E][F][D] -> W2t [E][D][F]
  transpose_cvt<<<dim3(D_FF / 64, D_MODEL / 64, NE), 256, 0, stream>>>(
      W1, W1t, D_MODEL, D_FF);
  transpose_cvt<<<dim3(D_MODEL / 64, D_FF / 64, NE), 256, 0, stream>>>(
      W2, W2t, D_FF, D_MODEL);

  moe_gate<<<NTOK / 8, 256, 0, stream>>>(x, gw, gb, counts, tokIdx,
                                         slotE, slotS, slotW);
  scan_counts<<<1, 32, 0, stream>>>(counts, offsets);

  dim3 g1(NTOK / MTA, NE);
  moe_ffn1<<<g1, 256, 0, stream>>>(xb, W1t, b1, counts, tokIdx, offsets, hbuf);

  dim3 g2(NTOK / MTB, D_MODEL / DBLK, NE);
  moe_ffn2<<<g2, 256, 0, stream>>>(hbuf, W2t, counts, offsets, ybuf);

  moe_combine<<<(NTOK * (D_MODEL / 4)) / 256, 256, 0, stream>>>(
      ybuf, b2, slotE, slotS, slotW, offsets, out);
}
